// GeoConvNet3DPC_28570122453862
// MI455X (gfx1250) — matrix operand order repaired and checked by reading, not compiled, not run on hardware
//
#include <hip/hip_runtime.h>
#include <hip/hip_bf16.h>

// ---------------------------------------------------------------------------
// PointNet++ classification on MI455X (gfx1250).
// Dense MLP work (~140 GFLOP) runs through v_wmma_f32_16x16x32_f16 (f16 in,
// f32 accumulate). GEMM uses 2x2 register blocking (32x32 tile per wave):
// 4 WMMAs per K-step from 2 A-frags + 2 B-frags -> 16 FLOP/byte from L2
// (192 MB L2 holds the activations), 2:1 b128-load:wmma ratio.
// FPS is the serial latency chain (one block/batch, LDS-resident min-dist).
// Ball query = exact k-smallest within radius via wave32 key-selection.
// ---------------------------------------------------------------------------

typedef _Float16 v16h __attribute__((ext_vector_type(16)));
typedef _Float16 v8h  __attribute__((ext_vector_type(8)));
typedef float    v8f  __attribute__((ext_vector_type(8)));

#define BATCH 8

// ---------------------------------------------------------------------------
// FPS: one block per batch element. mind[] lives in LDS; block-wide argmax
// reduction each step. Matches reference: idx[0]=0; idx[s]=argmax after
// updating mind with pos[idx[s-1]]. Tie-break = lowest index.
// ---------------------------------------------------------------------------
__global__ void __launch_bounds__(256) fps_kernel(const float* __restrict__ pos,
                                                  int N, int m,
                                                  int* __restrict__ idx) {
  __shared__ float mind[4096];
  __shared__ float rv[256];
  __shared__ int   ri[256];
  const int b = blockIdx.x;
  const int t = threadIdx.x;
  const float* p = pos + (size_t)b * N * 3;
  for (int i = t; i < N; i += 256) mind[i] = 3.4e38f;
  if (t == 0) idx[b * m] = 0;
  __syncthreads();
  int cur = 0;
  for (int s = 1; s < m; ++s) {
    const float cx = p[cur * 3 + 0], cy = p[cur * 3 + 1], cz = p[cur * 3 + 2];
    float bv = -1.0f; int bi = 0x7fffffff;
    for (int i = t; i < N; i += 256) {
      const float dx = p[i * 3 + 0] - cx;
      const float dy = p[i * 3 + 1] - cy;
      const float dz = p[i * 3 + 2] - cz;
      const float d  = dx * dx + dy * dy + dz * dz;
      const float mv = fminf(mind[i], d);
      mind[i] = mv;
      if (mv > bv) { bv = mv; bi = i; }
    }
    rv[t] = bv; ri[t] = bi;
    __syncthreads();
    for (int stp = 128; stp > 0; stp >>= 1) {
      if (t < stp) {
        const float ov = rv[t + stp]; const int oi = ri[t + stp];
        if (ov > rv[t] || (ov == rv[t] && oi < ri[t])) { rv[t] = ov; ri[t] = oi; }
      }
      __syncthreads();
    }
    cur = ri[0];
    if (t == 0) idx[b * m + s] = cur;
    __syncthreads();
  }
}

// Gather sampled centroid coordinates.
__global__ void gather_cent(const float* __restrict__ pos,
                            const int* __restrict__ idx,
                            float* __restrict__ cent, int N, int m) {
  const int t = blockIdx.x * blockDim.x + threadIdx.x;
  if (t >= BATCH * m) return;
  const int b = t / m;
  const int id = idx[t];
  const float* p = pos + ((size_t)b * N + id) * 3;
  cent[t * 3 + 0] = p[0];
  cent[t * 3 + 1] = p[1];
  cent[t * 3 + 2] = p[2];
}

// ---------------------------------------------------------------------------
// Ball query: k nearest points within radius^2 per centroid. One wave32 per
// centroid. Distances cached in LDS (each lane only re-reads its own stride),
// then k rounds of "min 64-bit key strictly greater than previous", reduced
// across the wave with shfl_xor. key = (d2_bits << 32) | (idx+1).
// ---------------------------------------------------------------------------
__global__ void ball_query(const float* __restrict__ pos,
                           const float* __restrict__ cent,
                           int* __restrict__ nidx,
                           int N, int m, int k, float r2) {
  extern __shared__ float d2s[];
  const int wave = threadIdx.x >> 5;
  const int lane = threadIdx.x & 31;
  const int gc = blockIdx.x * (blockDim.x >> 5) + wave;
  if (gc >= BATCH * m) return;
  const int b = gc / m;
  float* d2 = d2s + (size_t)wave * N;
  const float* p = pos + (size_t)b * N * 3;
  const float cx = cent[gc * 3 + 0], cy = cent[gc * 3 + 1], cz = cent[gc * 3 + 2];
  for (int c = lane; c < N; c += 32) {
    const float dx = p[c * 3 + 0] - cx;
    const float dy = p[c * 3 + 1] - cy;
    const float dz = p[c * 3 + 2] - cz;
    d2[c] = dx * dx + dy * dy + dz * dz;
  }
  unsigned long long prev = 0ull;
  const size_t out_base = (size_t)gc * k;
  for (int j = 0; j < k; ++j) {
    unsigned long long best = ~0ull;
    for (int c = lane; c < N; c += 32) {
      const float d = d2[c];
      if (d <= r2) {
        const unsigned long long key =
            ((unsigned long long)__float_as_uint(d) << 32) | (unsigned)(c + 1);
        if (key > prev && key < best) best = key;
      }
    }
    for (int o = 16; o > 0; o >>= 1) {
      const unsigned long long other = __shfl_xor(best, o, 32);
      if (other < best) best = other;
    }
    if (lane == 0)
      nidx[out_base + j] = (best == ~0ull) ? -1 : (int)((best & 0xffffffffu) - 1u);
    prev = best;
  }
}

// ---------------------------------------------------------------------------
// Build f16 feature rows [x_j (Cx) | pos_j - cent (3) | zero pad] (Kpad cols).
// ---------------------------------------------------------------------------
__global__ void build_feat(const float* __restrict__ pos_src,
                           const float* __restrict__ x_src,
                           const float* __restrict__ cent,
                           const int* __restrict__ nidx,
                           _Float16* __restrict__ feat,
                           int N, int m, int k, int Cx, int Kpad) {
  const int r = blockIdx.x * blockDim.x + threadIdx.x;
  if (r >= BATCH * m * k) return;
  const int gc = r / k;
  const int b = gc / m;
  const int nj = nidx[r];
  _Float16* f = feat + (size_t)r * Kpad;
  if (nj < 0) {
    for (int c = 0; c < Kpad; ++c) f[c] = (_Float16)0.0f;
    return;
  }
  const float* pj = pos_src + ((size_t)b * N + nj) * 3;
  const float rx = pj[0] - cent[gc * 3 + 0];
  const float ry = pj[1] - cent[gc * 3 + 1];
  const float rz = pj[2] - cent[gc * 3 + 2];
  int c = 0;
  if (Cx > 0) {
    const float* xj = x_src + ((size_t)b * N + nj) * Cx;
    for (; c < Cx; ++c) f[c] = (_Float16)xj[c];
  }
  f[c++] = (_Float16)rx;
  f[c++] = (_Float16)ry;
  f[c++] = (_Float16)rz;
  for (; c < Kpad; ++c) f[c] = (_Float16)0.0f;
}

// ---------------------------------------------------------------------------
// Weight prep: W (Korig x N f32, row-major) -> Wt (N x Kpad f16, row-major,
// zero-padded K), plus folded BN scale/shift.
// ---------------------------------------------------------------------------
__global__ void prep_weights(const float* __restrict__ W,
                             const float* __restrict__ g,
                             const float* __restrict__ bb,
                             const float* __restrict__ mu,
                             const float* __restrict__ var,
                             _Float16* __restrict__ Wt,
                             float* __restrict__ scale,
                             float* __restrict__ shift,
                             int Korig, int Kpad, int N) {
  const int t = blockIdx.x * blockDim.x + threadIdx.x;
  if (t < N) {
    const float s = g[t] * rsqrtf(var[t] + 1e-5f);
    scale[t] = s;
    shift[t] = bb[t] - mu[t] * s;
  }
  if (t >= N * Kpad) return;
  const int n = t / Kpad, kk = t % Kpad;
  Wt[t] = (kk < Korig) ? (_Float16)W[(size_t)kk * N + n] : (_Float16)0.0f;
}

// ---------------------------------------------------------------------------
// WMMA GEMM + BN + ReLU:  Out(MxN,f16) = relu( A(MxK,f16) * W + BN ).
// Wt is W transposed (N x K). One wave per 32x32 output tile (2x2 blocking of
// 16x16 WMMA tiles), 8 waves/block. K mult of 32, M and N mult of 32.
//   A frag (16-bit 16x32): lane L (m=L&15, grp=L>>4):
//     halfs 0..7  = A[m][kk + grp*8 + i],  halfs 8..15 = A[m][kk+16+grp*8+i]
//   B frag (16-bit 32x16): lane L (n=L&15, grp=L>>4):
//     halfs 0..7  = Wt[n][kk + grp*8 + i],  halfs 8..15 = Wt[n][kk+16+grp*8+i]
//   C/D f32 16x16: acc[v] = D[grp*8 + v][L&15] within the 16x16 tile
// ---------------------------------------------------------------------------
__global__ void __launch_bounds__(256)
gemm_bn_relu(const _Float16* __restrict__ A,
             const _Float16* __restrict__ Wt,
             const float* __restrict__ scale,
             const float* __restrict__ shift,
             _Float16* __restrict__ Out,
             int M, int K, int N) {
  const int lane = threadIdx.x & 31;
  const int wave = threadIdx.x >> 5;
  const int tilesN = N >> 5;                 // 32-wide tiles
  const int tiles = (M >> 5) * tilesN;
  const int gw = blockIdx.x * 8 + wave;
  if (gw >= tiles) return;
  const int tm = gw / tilesN;
  const int tn = gw % tilesN;
  const int lm = lane & 15;
  const int g  = lane >> 4;

  const _Float16* arow0 = A  + ((size_t)(tm * 32 + lm)) * K + g * 8;
  const _Float16* arow1 = arow0 + (size_t)16 * K;
  const _Float16* brow0 = Wt + ((size_t)(tn * 32 + lm)) * K + g * 8;
  const _Float16* brow1 = brow0 + (size_t)16 * K;

  v8f acc00 = {0.f,0.f,0.f,0.f,0.f,0.f,0.f,0.f};
  v8f acc01 = {0.f,0.f,0.f,0.f,0.f,0.f,0.f,0.f};
  v8f acc10 = {0.f,0.f,0.f,0.f,0.f,0.f,0.f,0.f};
  v8f acc11 = {0.f,0.f,0.f,0.f,0.f,0.f,0.f,0.f};
  union H { v16h v; v8h h[2]; };
  H a0, a1, b0, b1;

  for (int kk = 0; kk < K; kk += 32) {
    a0.h[0] = *(const v8h*)(arow0 + kk);
    a0.h[1] = *(const v8h*)(arow0 + kk + 16);
    a1.h[0] = *(const v8h*)(arow1 + kk);
    a1.h[1] = *(const v8h*)(arow1 + kk + 16);
    b0.h[0] = *(const v8h*)(brow0 + kk);
    b0.h[1] = *(const v8h*)(brow0 + kk + 16);
    b1.h[0] = *(const v8h*)(brow1 + kk);
    b1.h[1] = *(const v8h*)(brow1 + kk + 16);
    acc00 = __builtin_amdgcn_wmma_f32_16x16x32_f16(
        false, a0.v, false, b0.v, (short)0, acc00, false, false);
    acc01 = __builtin_amdgcn_wmma_f32_16x16x32_f16(
        false, a0.v, false, b1.v, (short)0, acc01, false, false);
    acc10 = __builtin_amdgcn_wmma_f32_16x16x32_f16(
        false, a1.v, false, b0.v, (short)0, acc10, false, false);
    acc11 = __builtin_amdgcn_wmma_f32_16x16x32_f16(
        false, a1.v, false, b1.v, (short)0, acc11, false, false);
  }

  const int n0 = tn * 32 + lm;
  const int n1 = n0 + 16;
  const float sc0 = scale[n0], sh0 = shift[n0];
  const float sc1 = scale[n1], sh1 = shift[n1];
  const int r0 = tm * 32 + g * 8;
  const int r1 = r0 + 16;
#pragma unroll
  for (int v = 0; v < 8; ++v) {
    const size_t row0 = (size_t)(r0 + v) * N;
    const size_t row1 = (size_t)(r1 + v) * N;
    Out[row0 + n0] = (_Float16)fmaxf(acc00[v] * sc0 + sh0, 0.0f);
    Out[row0 + n1] = (_Float16)fmaxf(acc01[v] * sc1 + sh1, 0.0f);
    Out[row1 + n0] = (_Float16)fmaxf(acc10[v] * sc0 + sh0, 0.0f);
    Out[row1 + n1] = (_Float16)fmaxf(acc11[v] * sc1 + sh1, 0.0f);
  }
}

// Max over k neighbors (invalid slots skipped, like reference's -inf mask).
__global__ void max_aggr(const _Float16* __restrict__ h,
                         const int* __restrict__ nidx,
                         float* __restrict__ out,
                         int bm, int k, int C) {
  const int t = blockIdx.x * blockDim.x + threadIdx.x;
  if (t >= bm * C) return;
  const int rc = t / C, c = t % C;
  float acc = -3.4e38f;
  const int* nb = nidx + (size_t)rc * k;
  const _Float16* hr = h + (size_t)rc * k * C + c;
  for (int j = 0; j < k; ++j)
    if (nb[j] >= 0) acc = fmaxf(acc, (float)hr[(size_t)j * C]);
  out[t] = acc;
}

// Global max pool over m points -> f16, padded to 32 rows for WMMA head.
__global__ void global_pool_pad32(const float* __restrict__ x,
                                  _Float16* __restrict__ out, int m, int C) {
  const int t = blockIdx.x * blockDim.x + threadIdx.x;
  if (t >= 32 * C) return;
  const int b = t / C, c = t % C;
  if (b >= BATCH) { out[t] = (_Float16)0.0f; return; }
  float acc = -3.4e38f;
  for (int i = 0; i < m; ++i)
    acc = fmaxf(acc, x[((size_t)b * m + i) * C + c]);
  out[t] = (_Float16)acc;
}

// Final 256 -> 40 classifier (tiny; scalar).
__global__ void final_linear(const _Float16* __restrict__ h,
                             const float* __restrict__ W3,
                             const float* __restrict__ b3,
                             float* __restrict__ out) {
  const int t = blockIdx.x * blockDim.x + threadIdx.x;
  if (t >= BATCH * 40) return;
  const int b = t / 40, o = t % 40;
  float acc = b3[o];
  for (int c = 0; c < 256; ++c)
    acc += (float)h[b * 256 + c] * W3[c * 40 + o];
  out[t] = acc;
}

// ---------------------------------------------------------------------------
extern "C" void kernel_launch(void* const* d_in, const int* in_sizes, int n_in,
                              void* d_out, int out_size, void* d_ws, size_t ws_size,
                              hipStream_t stream) {
  (void)in_sizes; (void)n_in; (void)out_size; (void)ws_size;

  const float* pos = (const float*)d_in[0];

  // ---- workspace carve (deterministic) ----
  char* base = (char*)d_ws;
  size_t off = 0;
  auto alloc = [&](size_t bytes) -> void* {
    void* p = base + off;
    off += (bytes + 255) & ~(size_t)255;
    return p;
  };

  int*   idx1 = (int*)  alloc((size_t)BATCH * 2048 * 4);
  int*   idx2 = (int*)  alloc((size_t)BATCH * 512  * 4);
  int*   idx3 = (int*)  alloc((size_t)BATCH * 64   * 4);
  float* cent1= (float*)alloc((size_t)BATCH * 2048 * 3 * 4);
  float* cent2= (float*)alloc((size_t)BATCH * 512  * 3 * 4);
  float* cent3= (float*)alloc((size_t)BATCH * 64   * 3 * 4);
  int*   nidx1= (int*)  alloc((size_t)BATCH * 2048 * 32  * 4);
  int*   nidx2= (int*)  alloc((size_t)BATCH * 512  * 64  * 4);
  int*   nidx3= (int*)  alloc((size_t)BATCH * 64   * 128 * 4);
  float* sa1o = (float*)alloc((size_t)BATCH * 2048 * 128  * 4);
  float* sa2o = (float*)alloc((size_t)BATCH * 512  * 256  * 4);
  float* sa3o = (float*)alloc((size_t)BATCH * 64   * 1024 * 4);
  _Float16* pooled = (_Float16*)alloc((size_t)32 * 1024 * 2);
  _Float16* h1 = (_Float16*)alloc((size_t)32 * 512 * 2);
  _Float16* h2 = (_Float16*)alloc((size_t)32 * 256 * 2);
  _Float16* feat = (_Float16*)alloc((size_t)262144 * 160 * 2);   // max feat buf
  _Float16* actA = (_Float16*)alloc((size_t)524288 * 128 * 2);   // layer 1/3 out
  _Float16* actB = (_Float16*)alloc((size_t)524288 * 64  * 2);   // layer 2 out

  // prepped weights (Wt f16 NxKpad) + BN scale/shift
  struct Layer { _Float16* Wt; float* sc; float* sh; int Ko, Kp, N; int in_idx; };
  Layer L[11] = {
      {0,0,0,   3,  32,   64,  1}, {0,0,0,  64,  64,   64,  6}, {0,0,0,  64,  64,  128, 11},
      {0,0,0, 131, 160,  128, 16}, {0,0,0, 128, 128,  128, 21}, {0,0,0, 128, 128,  256, 26},
      {0,0,0, 259, 288,  256, 31}, {0,0,0, 256, 256,  512, 36}, {0,0,0, 512, 512, 1024, 41},
      {0,0,0,1024,1024,  512, 46}, {0,0,0, 512, 512,  256, 51}};
  for (int i = 0; i < 11; ++i) {
    L[i].Wt = (_Float16*)alloc((size_t)L[i].N * L[i].Kp * 2);
    L[i].sc = (float*)   alloc((size_t)L[i].N * 4);
    L[i].sh = (float*)   alloc((size_t)L[i].N * 4);
    const int total = L[i].N * L[i].Kp;
    prep_weights<<<(total + 255) / 256, 256, 0, stream>>>(
        (const float*)d_in[L[i].in_idx],     (const float*)d_in[L[i].in_idx + 1],
        (const float*)d_in[L[i].in_idx + 2], (const float*)d_in[L[i].in_idx + 3],
        (const float*)d_in[L[i].in_idx + 4],
        L[i].Wt, L[i].sc, L[i].sh, L[i].Ko, L[i].Kp, L[i].N);
  }

  auto gemm = [&](const _Float16* A, int li, _Float16* O, int M) {
    const int tiles = (M / 32) * (L[li].N / 32);
    gemm_bn_relu<<<(tiles + 7) / 8, 256, 0, stream>>>(
        A, L[li].Wt, L[li].sc, L[li].sh, O, M, L[li].Kp, L[li].N);
  };

  // ================= SA1: N=4096 -> m=2048, r=0.2, k=32 =================
  {
    const int N = 4096, m = 2048, k = 32;
    const float r2 = 0.2f * 0.2f;
    fps_kernel<<<BATCH, 256, 0, stream>>>(pos, N, m, idx1);
    gather_cent<<<(BATCH * m + 255) / 256, 256, 0, stream>>>(pos, idx1, cent1, N, m);
    ball_query<<<(BATCH * m + 3) / 4, 128, 4 * N * sizeof(float), stream>>>(
        pos, cent1, nidx1, N, m, k, r2);
    const int rows = BATCH * m * k;  // 524288
    build_feat<<<(rows + 255) / 256, 256, 0, stream>>>(
        pos, nullptr, cent1, nidx1, feat, N, m, k, 0, 32);
    gemm(feat, 0, actA, rows);   // 32  -> 64
    gemm(actA, 1, actB, rows);   // 64  -> 64
    gemm(actB, 2, actA, rows);   // 64  -> 128
    max_aggr<<<(BATCH * m * 128 + 255) / 256, 256, 0, stream>>>(
        actA, nidx1, sa1o, BATCH * m, k, 128);
  }

  // ================= SA2: N=2048 -> m=512, r=0.4, k=64 =================
  {
    const int N = 2048, m = 512, k = 64;
    const float r2 = 0.4f * 0.4f;
    fps_kernel<<<BATCH, 256, 0, stream>>>(cent1, N, m, idx2);
    gather_cent<<<(BATCH * m + 255) / 256, 256, 0, stream>>>(cent1, idx2, cent2, N, m);
    ball_query<<<(BATCH * m + 3) / 4, 128, 4 * N * sizeof(float), stream>>>(
        cent1, cent2, nidx2, N, m, k, r2);
    const int rows = BATCH * m * k;  // 262144
    build_feat<<<(rows + 255) / 256, 256, 0, stream>>>(
        cent1, sa1o, cent2, nidx2, feat, N, m, k, 128, 160);
    gemm(feat, 3, actA, rows);   // 160 -> 128
    gemm(actA, 4, actB, rows);   // 128 -> 128
    gemm(actB, 5, actA, rows);   // 128 -> 256
    max_aggr<<<(BATCH * m * 256 + 255) / 256, 256, 0, stream>>>(
        actA, nidx2, sa2o, BATCH * m, k, 256);
  }

  // ================= SA3: N=512 -> m=64, r=1e6 (all), k=128 =================
  {
    const int N = 512, m = 64, k = 128;
    const float r2 = 1e12f;
    fps_kernel<<<BATCH, 256, 0, stream>>>(cent2, N, m, idx3);
    gather_cent<<<(BATCH * m + 255) / 256, 256, 0, stream>>>(cent2, idx3, cent3, N, m);
    ball_query<<<(BATCH * m + 3) / 4, 128, 4 * N * sizeof(float), stream>>>(
        cent2, cent3, nidx3, N, m, k, r2);
    const int rows = BATCH * m * k;  // 65536
    build_feat<<<(rows + 255) / 256, 256, 0, stream>>>(
        cent2, sa2o, cent3, nidx3, feat, N, m, k, 256, 288);
    gemm(feat, 6, actA, rows);   // 288 -> 256
    gemm(actA, 7, actB, rows);   // 256 -> 512
    gemm(actB, 8, actA, rows);   // 512 -> 1024
    max_aggr<<<(BATCH * m * 1024 + 255) / 256, 256, 0, stream>>>(
        actA, nidx3, sa3o, BATCH * m, k, 1024);
  }

  // ================= Head =================
  global_pool_pad32<<<(32 * 1024 + 255) / 256, 256, 0, stream>>>(sa3o, pooled, 64, 1024);
  gemm(pooled, 9,  h1, 32);   // 1024 -> 512 (+BN+ReLU)
  gemm(h1,    10,  h2, 32);   // 512  -> 256 (+BN+ReLU)
  final_linear<<<(BATCH * 40 + 63) / 64, 64, 0, stream>>>(
      h2, (const float*)d_in[56], (const float*)d_in[57], (float*)d_out);
}
